// MultiHeadAttention_17815524344555
// MI455X (gfx1250) — matrix operand order repaired and checked by reading, not compiled, not run on hardware
//
#include <hip/hip_runtime.h>
#include <hip/hip_bf16.h>
#include <math.h>

// ---------------- problem constants ----------------
#define BATCH 4
#define SEQ   2048
#define HEADS 16
#define DK    64
#define DM    1024
#define MROWS (BATCH * SEQ)      // 8192
#define NEGV  (-1000000000.0f)

// ---------------- WMMA types ----------------
typedef __bf16 v16bf __attribute__((ext_vector_type(16)));
typedef float  v8f   __attribute__((ext_vector_type(8)));
typedef unsigned int u32x4 __attribute__((ext_vector_type(4)));

static __device__ __forceinline__ __bf16 f2bf(float f) {
    union { float f; unsigned int u; } v; v.f = f;
    unsigned int r = v.u + 0x7FFFu + ((v.u >> 16) & 1u);   // RNE
    unsigned short h = (unsigned short)(r >> 16);
    __bf16 o;
    __builtin_memcpy(&o, &h, 2);
    return o;
}

static __device__ __forceinline__ v8f wmma_bf16(v16bf a, v16bf b, v8f c) {
    return __builtin_amdgcn_wmma_f32_16x16x32_bf16(
        /*neg_a=*/false, a, /*neg_b=*/false, b,
        /*c_mod=*/(short)0, c, /*reuse_a=*/false, /*reuse_b=*/false);
}

// A-matrix 16x32 bf16 fragment, source row-major [16][stride] (K contiguous in a row).
static __device__ __forceinline__ v16bf frag_a_rowmajor(const __bf16* base, int stride, int lane) {
    int r  = lane & 15;
    int hi = lane >> 4;
    const __bf16* p = base + (size_t)r * stride + hi * 8;
    union { v16bf v; u32x4 q[2]; } f;
    f.q[0] = *(const u32x4*)(p);        // K: hi*8 + 0..7
    f.q[1] = *(const u32x4*)(p + 16);   // K: 16 + hi*8 + 0..7
    return f.v;
}

// A-matrix 16x32 fragment from fp32 row-major storage, converting to bf16 in-register.
static __device__ __forceinline__ v16bf frag_a_f32(const float* base, int stride, int lane) {
    int r  = lane & 15;
    int hi = lane >> 4;
    const float* rp = base + (size_t)r * stride + hi * 8;
    float4 c0 = *(const float4*)(rp);
    float4 c1 = *(const float4*)(rp + 4);
    float4 c2 = *(const float4*)(rp + 16);
    float4 c3 = *(const float4*)(rp + 20);
    v16bf a;
    a[0]  = f2bf(c0.x); a[1]  = f2bf(c0.y); a[2]  = f2bf(c0.z); a[3]  = f2bf(c0.w);
    a[4]  = f2bf(c1.x); a[5]  = f2bf(c1.y); a[6]  = f2bf(c1.z); a[7]  = f2bf(c1.w);
    a[8]  = f2bf(c2.x); a[9]  = f2bf(c2.y); a[10] = f2bf(c2.z); a[11] = f2bf(c2.w);
    a[12] = f2bf(c3.x); a[13] = f2bf(c3.y); a[14] = f2bf(c3.z); a[15] = f2bf(c3.w);
    return a;
}

// B-matrix 32x16 bf16 fragment from storage bt[n][k] (K contiguous per n-row).
static __device__ __forceinline__ v16bf frag_b_kcontig(const __bf16* base, int stride, int lane) {
    int n  = lane & 15;
    int hi = lane >> 4;
    const __bf16* p = base + (size_t)n * stride + hi * 8;
    union { v16bf v; u32x4 q[2]; } f;
    f.q[0] = *(const u32x4*)(p);
    f.q[1] = *(const u32x4*)(p + 16);
    return f.v;
}

static __device__ __forceinline__ unsigned lds_off(const void* p) {
    // generic LDS pointer: low 32 bits are the LDS byte offset (ISA 10.2 aperture rules)
    return (unsigned)(unsigned long long)p;
}

// =====================================================================
// Generic GEMM  Y = X(MxK,f32) * W(KxN,f32) + bias, bf16 WMMA, f32 acc.
// A tile staged fp32 via GLOBAL_LOAD_ASYNC_TO_LDS_B128 (ASYNCcnt),
// double-buffered so the copy of tile it+1 overlaps WMMA of tile it.
// B tile register-pipelined + converted to bf16, stored transposed.
// MODE 0: outF[row*N+col] = y                       (final projection)
// MODE 1: outH[((b*H+h)*SEQ+s)*DK+d] = bf16(y)      (q,k head layout)
// MODE 2: outH[((b*H+h)*DK+d)*SEQ+s] = bf16(y)      (v transposed)
// Block: 256 thr (8 waves), tile BM=128 BN=64 BK=32; wave = 32x32 (2x2 WMMA)
// =====================================================================
#define BM 128
#define BN 64
#define BK 32

static __device__ __forceinline__ void issue_asyncA(const float* X, int bm, int k0, int K,
                                                    unsigned ldsbase, int tid) {
#pragma unroll
    for (int i = 0; i < 4; ++i) {
        int slot = tid + 256 * i;            // 0..1023 float4 slots of 128x32 tile
        int r = slot >> 3;                   // 8 float4 per 32-wide row
        int c = (slot & 7) << 2;
        const float* g = X + (size_t)(bm + r) * K + k0 + c;
        unsigned l = ldsbase + (unsigned)(r * BK + c) * 4u;
        asm volatile("global_load_async_to_lds_b128 %0, %1, off"
                     :: "v"(l), "v"(g) : "memory");
    }
}

static __device__ __forceinline__ void loadB_regs(const float* W, int k0, int bn, int N,
                                                  int tid, float4& w0, float4& w1) {
    int s0 = tid, s1 = tid + 256;            // 512 float4 slots of 32x64 tile
    w0 = *(const float4*)(W + (size_t)(k0 + (s0 >> 4)) * N + bn + ((s0 & 15) << 2));
    w1 = *(const float4*)(W + (size_t)(k0 + (s1 >> 4)) * N + bn + ((s1 & 15) << 2));
}

static __device__ __forceinline__ void storeB_lds(__bf16* bt, int tid, float4 w0, float4 w1) {
    int r0 = tid >> 4,        c0 = (tid & 15) << 2;
    int r1 = (tid + 256) >> 4, c1 = ((tid + 256) & 15) << 2;
    bt[(c0 + 0) * BK + r0] = f2bf(w0.x);
    bt[(c0 + 1) * BK + r0] = f2bf(w0.y);
    bt[(c0 + 2) * BK + r0] = f2bf(w0.z);
    bt[(c0 + 3) * BK + r0] = f2bf(w0.w);
    bt[(c1 + 0) * BK + r1] = f2bf(w1.x);
    bt[(c1 + 1) * BK + r1] = f2bf(w1.y);
    bt[(c1 + 2) * BK + r1] = f2bf(w1.z);
    bt[(c1 + 3) * BK + r1] = f2bf(w1.w);
}

template<int MODE>
__global__ __launch_bounds__(256)
void gemm_bias_kernel(const float* __restrict__ X, const float* __restrict__ W,
                      const float* __restrict__ bias,
                      float* __restrict__ outF, __bf16* __restrict__ outH,
                      int M, int K, int N) {
    __shared__ float  ldsAf[2][BM * BK];   // fp32 row-major [128][32], 2x16KB
    __shared__ __bf16 ldsBt[2][BN * BK];   // bf16 transposed [n=64][k=32], 2x4KB

    const int tid  = threadIdx.x;
    const int lane = tid & 31;
    const int w    = tid >> 5;
    const int bm   = blockIdx.x * BM;
    const int bn   = blockIdx.y * BN;
    const int wm   = (w >> 1) * 32;        // 4 wave rows
    const int wn   = (w & 1) * 32;         // 2 wave cols

    unsigned aoff[2] = { lds_off(&ldsAf[0][0]), lds_off(&ldsAf[1][0]) };

    v8f acc[2][2] = {};
    float4 wr0, wr1;

    // prologue: tile 0 in flight
    issue_asyncA(X, bm, 0, K, aoff[0], tid);
    loadB_regs(W, 0, bn, N, tid, wr0, wr1);

    const int nIter = K / BK;
    for (int it = 0; it < nIter; ++it) {
        const int k0  = it * BK;
        const int buf = it & 1;

        // convert+store current B tile (regs loaded last iteration)
        storeB_lds(&ldsBt[buf][0], tid, wr0, wr1);
        // wait for this buffer's async A copies (issued last iteration)
        asm volatile("s_wait_asynccnt 0x0" ::: "memory");
        __syncthreads();
        // now safe to start filling the other buffer: every wave passed the
        // barrier, so all reads of buf^1 (previous iteration) are complete
        if (it + 1 < nIter) {
            issue_asyncA(X, bm, k0 + BK, K, aoff[buf ^ 1], tid);
            loadB_regs(W, k0 + BK, bn, N, tid, wr0, wr1);
        }

        const float*  Af = &ldsAf[buf][0];
        const __bf16* Bt = &ldsBt[buf][0];
        v16bf af[2], bfr[2];
        af[0]  = frag_a_f32(Af + (size_t)(wm +  0) * BK, BK, lane);
        af[1]  = frag_a_f32(Af + (size_t)(wm + 16) * BK, BK, lane);
        bfr[0] = frag_b_kcontig(Bt + (size_t)(wn +  0) * BK, BK, lane);
        bfr[1] = frag_b_kcontig(Bt + (size_t)(wn + 16) * BK, BK, lane);
#pragma unroll
        for (int mi = 0; mi < 2; ++mi)
#pragma unroll
            for (int ni = 0; ni < 2; ++ni)
                acc[mi][ni] = wmma_bf16(af[mi], bfr[ni], acc[mi][ni]);
        // single barrier per iteration is sufficient for the double buffer:
        // next iteration's writes hit buf^1, last read two barriers ago.
    }

    // epilogue
#pragma unroll
    for (int mi = 0; mi < 2; ++mi) {
#pragma unroll
        for (int ni = 0; ni < 2; ++ni) {
            int colg = bn + wn + 16 * ni + (lane & 15);
            float bv = bias[colg];
#pragma unroll
            for (int j = 0; j < 8; ++j) {
                int rowg = bm + wm + 16 * mi + 8 * (lane >> 4) + j;
                float y = acc[mi][ni][j] + bv;
                if (MODE == 0) {
                    outF[(size_t)rowg * N + colg] = y;
                } else {
                    int b = rowg >> 11, s = rowg & (SEQ - 1);
                    int h = colg >> 6,  d = colg & (DK - 1);
                    if (MODE == 1)
                        outH[(((size_t)(b * HEADS + h) * SEQ) + s) * DK + d] = f2bf(y);
                    else
                        outH[(((size_t)(b * HEADS + h) * DK) + d) * SEQ + s] = f2bf(y);
                }
            }
        }
    }
}

// =====================================================================
// Fused attention: per block = (b,h, 32 q-rows)
//   scores(32x2048) = q(32x64) @ k^T : WMMA, scale, +rel, mask -> LDS
//   softmax rows in LDS, attn written to global once
//   out(32x64) = attn(bf16 on the fly) @ v  via vT
// LDS: 32*2048*4 = 256 KB dynamic (fits 320 KB/WGP, 1 WG per WGP)
// =====================================================================
__global__ __launch_bounds__(256)
void attn_kernel(const __bf16* __restrict__ qh, const __bf16* __restrict__ kh,
                 const __bf16* __restrict__ vT, const int* __restrict__ mask,
                 const float* __restrict__ rel,
                 float* __restrict__ attn_out, float* __restrict__ concat) {
    extern __shared__ float sc[];                 // [32][SEQ]
    const int tid  = threadIdx.x;
    const int lane = tid & 31;
    const int w    = tid >> 5;
    const int bh   = blockIdx.y;                  // b*HEADS + h
    const int b    = bh >> 4, h = bh & 15;
    const int q0   = blockIdx.x * 32;

    const __bf16* qbase = qh + ((size_t)bh * SEQ + q0) * DK;
    const __bf16* kbase = kh + (size_t)bh * SEQ * DK;

    // ---- phase 1: scores ----
    {
        const int mi  = w & 1;                    // which 16-row half
        const int ntw = w >> 1;                   // 0..3 column group
        v16bf aq0 = frag_a_rowmajor(qbase + (size_t)mi * 16 * DK +  0, DK, lane);
        v16bf aq1 = frag_a_rowmajor(qbase + (size_t)mi * 16 * DK + 32, DK, lane);
        for (int t = 0; t < 32; ++t) {
            int n0 = (ntw + 4 * t) * 16;
            v16bf bk0 = frag_b_kcontig(kbase + (size_t)n0 * DK +  0, DK, lane);
            v16bf bk1 = frag_b_kcontig(kbase + (size_t)n0 * DK + 32, DK, lane);
            v8f s = {};
            s = wmma_bf16(aq0, bk0, s);
            s = wmma_bf16(aq1, bk1, s);
            int kcol = n0 + (lane & 15);
            bool dead = (mask[b * SEQ + kcol] == 0);
            const float* relp = rel + (((size_t)h * SEQ) + (q0 + mi * 16 + 8 * (lane >> 4))) * SEQ + kcol;
#pragma unroll
            for (int j = 0; j < 8; ++j) {
                int mrow = mi * 16 + 8 * (lane >> 4) + j;
                float val = s[j] * 0.125f + relp[(size_t)j * SEQ];  // 1/sqrt(64)
                if (dead) val = NEGV;
                sc[mrow * SEQ + kcol] = val;
            }
        }
    }
    __syncthreads();

    // ---- phase 2: softmax (wave w owns rows 4w..4w+3), write attn ----
    for (int rr = 0; rr < 4; ++rr) {
        int row = w * 4 + rr;
        float* rowp = sc + (size_t)row * SEQ;
        float mx = -INFINITY;
        for (int c = lane; c < SEQ; c += 32) mx = fmaxf(mx, rowp[c]);
#pragma unroll
        for (int off = 16; off; off >>= 1) mx = fmaxf(mx, __shfl_xor(mx, off));
        float sum = 0.0f;
        for (int c = lane; c < SEQ; c += 32) {
            float e = __expf(rowp[c] - mx);
            rowp[c] = e;
            sum += e;
        }
#pragma unroll
        for (int off = 16; off; off >>= 1) sum += __shfl_xor(sum, off);
        float inv = 1.0f / sum;
        float* gout = attn_out + ((size_t)bh * SEQ + q0 + row) * SEQ;
        for (int c = lane; c < SEQ; c += 32) {
            float p = rowp[c] * inv;
            rowp[c] = p;
            gout[c] = p;
        }
    }
    __syncthreads();

    // ---- phase 3: out = attn @ v  (wave w -> 16x16 tile: mi=w>>2, ni=w&3) ----
    {
        const int mi = w >> 2, ni = w & 3;
        const __bf16* vbase = vT + ((size_t)bh * DK + ni * 16) * SEQ;
        v8f oacc = {};
        for (int kk = 0; kk < SEQ; kk += 32) {
            v16bf av = frag_a_f32(sc + (size_t)(mi * 16) * SEQ + kk, SEQ, lane);
            v16bf bv = frag_b_kcontig(vbase + kk, SEQ, lane);
            oacc = wmma_bf16(av, bv, oacc);
        }
#pragma unroll
        for (int j = 0; j < 8; ++j) {
            int srow = q0 + mi * 16 + 8 * (lane >> 4) + j;
            int col  = h * DK + ni * 16 + (lane & 15);
            concat[((size_t)b * SEQ + srow) * DM + col] = oacc[j];
        }
    }
}

// =====================================================================
extern "C" void kernel_launch(void* const* d_in, const int* in_sizes, int n_in,
                              void* d_out, int out_size, void* d_ws, size_t ws_size,
                              hipStream_t stream) {
    const float* Q    = (const float*)d_in[0];
    const float* Kin  = (const float*)d_in[1];
    const float* V    = (const float*)d_in[2];
    const int*   mask = (const int*)  d_in[3];
    const float* rel  = (const float*)d_in[4];
    const float* Wq   = (const float*)d_in[5];
    const float* bq   = (const float*)d_in[6];
    const float* Wk   = (const float*)d_in[7];
    const float* bk   = (const float*)d_in[8];
    const float* Wv   = (const float*)d_in[9];
    const float* bv   = (const float*)d_in[10];
    const float* Wo   = (const float*)d_in[11];
    const float* bo   = (const float*)d_in[12];

    const size_t headElems = (size_t)BATCH * HEADS * SEQ * DK;   // 8,388,608
    __bf16* qhp = (__bf16*)d_ws;
    __bf16* khp = qhp + headElems;
    __bf16* vTp = khp + headElems;
    float*  concat = (float*)(vTp + headElems);                  // 8192x1024 f32

    float* outp  = (float*)d_out;                                // [B,S,DM]
    float* attnp = outp + (size_t)BATCH * SEQ * DM;              // [B,H,S,S]

    dim3 gg(MROWS / BM, DM / BN);        // (64, 16)
    gemm_bias_kernel<1><<<gg, 256, 0, stream>>>(Q,   Wq, bq, nullptr, qhp, MROWS, DM, DM);
    gemm_bias_kernel<1><<<gg, 256, 0, stream>>>(Kin, Wk, bk, nullptr, khp, MROWS, DM, DM);
    gemm_bias_kernel<2><<<gg, 256, 0, stream>>>(V,   Wv, bv, nullptr, vTp, MROWS, DM, DM);

    dim3 ga(SEQ / 32, BATCH * HEADS);    // (64, 64)
    attn_kernel<<<ga, 256, 32 * SEQ * sizeof(float), stream>>>(
        qhp, khp, vTp, mask, rel, attnp, concat);

    gemm_bias_kernel<0><<<gg, 256, 0, stream>>>(concat, Wo, bo, outp, nullptr, MROWS, DM, DM);
}
